// detokenization_56332791054483
// MI455X (gfx1250) — matrix operand order repaired and checked by reading, not compiled, not run on hardware
//
#include <hip/hip_runtime.h>
#include <hip/hip_bf16.h>

typedef __attribute__((ext_vector_type(16))) _Float16 v16h;
typedef __attribute__((ext_vector_type(8)))  _Float16 v8h;
typedef __attribute__((ext_vector_type(8)))  float    v8f;

#define Bq 8
#define Nq 65536
#define Cq 64
#define Hq 384
#define BLK_PER_B 32
#define ROWS_PER_BLK (Nq / BLK_PER_B)   // 2048
#define SLAB 32
#define NSLABS (ROWS_PER_BLK / SLAB)    // 64

// LDS strides (in halves); all row starts 16B-aligned
#define LDW 80    // W1s / W3s / W2s / Xcs / ts rows (64 cols + pad)
#define LDX 400   // xouts / TprevT rows (384 cols + pad)
#define LDT 40    // attnT / vT rows (32 cols + pad)

#define EXP_SHIFT 8.0f

// ---- WMMA operand loaders (wave32 layouts per CDNA5 ISA 7.12.2) ----------
// A (f16, 16x32, MxK): lane m = L&15; VGPR0-3 hold K = (L>>4)*8 .. +7,
// VGPR4-7 hold K = 16 + (L>>4)*8 .. +7.  Row-major [M][K] storage.
__device__ __forceinline__ v16h ldsA(const _Float16* base, int ld, int m0,
                                     int k0, int lane) {
  const _Float16* p = base + (m0 + (lane & 15)) * ld + k0 + ((lane >> 4) << 3);
  v8h lo = *(const v8h*)(p);
  v8h hi = *(const v8h*)(p + 16);
  v16h a;
#pragma unroll
  for (int i = 0; i < 8; ++i) { a[i] = lo[i]; a[i + 8] = hi[i]; }
  return a;
}

// B (f16, 32x16, KxN): lane n = L&15; VGPR0-3 hold K = (L>>4)*8 .. +7,
// VGPR4-7 hold K = 16 + (L>>4)*8 .. +7.
// Storage is B-transposed (rows indexed by N, cols by K) -> two runs of 8 contiguous
// halves per lane.
__device__ __forceinline__ v16h ldsBt(const _Float16* base, int ld, int n0,
                                      int k0, int lane) {
  const _Float16* p = base + (n0 + (lane & 15)) * ld + k0 + ((lane >> 4) << 3);
  v8h lo = *(const v8h*)(p);
  v8h hi = *(const v8h*)(p + 16);
  v16h b;
#pragma unroll
  for (int i = 0; i < 8; ++i) { b[i] = lo[i]; b[i + 8] = hi[i]; }
  return b;
}

__device__ __forceinline__ v8f wmma16(v16h a, v16h b, v8f c) {
  return __builtin_amdgcn_wmma_f32_16x16x32_f16(false, a, false, b, (short)0,
                                                c, false, false);
}

__device__ __forceinline__ v8f zero8() {
  v8f z = {0.f, 0.f, 0.f, 0.f, 0.f, 0.f, 0.f, 0.f};
  return z;
}

__device__ __forceinline__ v8h cvt8(float4 x, float4 y) {
  v8h o;
  o[0] = (_Float16)x.x; o[1] = (_Float16)x.y;
  o[2] = (_Float16)x.z; o[3] = (_Float16)x.w;
  o[4] = (_Float16)y.x; o[5] = (_Float16)y.y;
  o[6] = (_Float16)y.z; o[7] = (_Float16)y.w;
  return o;
}

// ---------------------------------------------------------------------------
__global__ void detok_zero_ws(float* p, int n) {
  int i = blockIdx.x * 256 + threadIdx.x;
  if (i < n) p[i] = 0.f;
}

__global__ __launch_bounds__(256, 1) void detok_fused(
    const float* __restrict__ Xc, const float* __restrict__ Tprev,
    const float* __restrict__ W1, const float* __restrict__ W2,
    const float* __restrict__ b2, const float* __restrict__ W3,
    const float* __restrict__ b3, float* __restrict__ ws_out,
    float* __restrict__ ws_denom) {
  __shared__ alignas(16) _Float16 W1s[Hq * LDW];      // 61440 B
  __shared__ alignas(16) _Float16 W3s[Hq * LDW];      // 61440 B
  __shared__ alignas(16) _Float16 W2s[Cq * LDW];      // 10240 B
  __shared__ alignas(16) _Float16 TprevT[Cq * LDX];   // 51200 B
  __shared__ alignas(16) _Float16 Xcs[SLAB * LDW];    //  5120 B
  __shared__ alignas(16) _Float16 xouts[SLAB * LDX];  // 25600 B
  __shared__ alignas(16) _Float16 ts[SLAB * LDW];     //  5120 B
  __shared__ alignas(16) _Float16 attnT[Cq * LDT];    //  5120 B
  __shared__ alignas(16) _Float16 vT[Hq * LDT];       // 30720 B
  __shared__ float b2s[Cq];
  __shared__ float b3s[Hq];
  __shared__ float denomLDS[Cq];

  const int tid = threadIdx.x;
  const int wave = tid >> 5;
  const int lane = tid & 31;
  const int b = blockIdx.x >> 5;        // batch
  const int blk = blockIdx.x & 31;      // n-chunk within batch
  const int hbase = wave * 48;          // 3 H-tiles per wave

  // ---- stage weights into LDS as f16 (once per block) ----
  for (int i = tid; i < (Hq * Cq) / 8; i += 256) {  // 8 floats per iter
    int h = i >> 3, cg = (i & 7) * 8;
    const float4* s1 = (const float4*)(W1 + h * Cq + cg);
    *(v8h*)&W1s[h * LDW + cg] = cvt8(s1[0], s1[1]);
    const float4* s3 = (const float4*)(W3 + h * Cq + cg);
    *(v8h*)&W3s[h * LDW + cg] = cvt8(s3[0], s3[1]);
  }
  for (int i = tid; i < (Cq * Cq) / 8; i += 256) {
    int d = i >> 3, cg = (i & 7) * 8;
    const float4* s2 = (const float4*)(W2 + d * Cq + cg);
    *(v8h*)&W2s[d * LDW + cg] = cvt8(s2[0], s2[1]);
  }
  {  // transpose Tprev[b] (HxC) -> TprevT (CxH)
    const float* Tp = Tprev + (size_t)b * Hq * Cq;
    for (int i = tid; i < Hq * Cq; i += 256) {
      int h = i >> 6, c = i & 63;
      TprevT[c * LDX + h] = (_Float16)Tp[i];
    }
  }
  for (int i = tid; i < Hq; i += 256) b3s[i] = b3[i];
  if (tid < Cq) { b2s[tid] = b2[tid]; denomLDS[tid] = 0.f; }

  // ---- persistent numerator accumulators: out[cTile(4)][hTile(3)] ----
  v8f oacc[4][3];
#pragma unroll
  for (int i = 0; i < 4; ++i)
#pragma unroll
    for (int j = 0; j < 3; ++j) oacc[i][j] = zero8();

  const int n_start = blk * ROWS_PER_BLK;
  const int s_mt = wave >> 2;  // m-tile owned in t/scores stages
  const int s_ct = wave & 3;   // c-tile owned in t/scores stages
  const int ld_row = tid >> 3;         // Xc slab loader: row 0..31
  const int ld_cg = (tid & 7) * 8;     // and 8-col group

  // ---- software pipeline: prefetch slab 0 while weights are staged ----
  float4 pf0, pf1;
  {
    const float4* p =
        (const float4*)(Xc + ((size_t)b * Nq + n_start + ld_row) * Cq + ld_cg);
    pf0 = p[0];
    pf1 = p[1];
  }

  __syncthreads();  // weights + biases visible

  for (int s = 0; s < NSLABS; ++s) {
    // commit prefetched Xc slab (f32 regs -> f16 LDS)
    *(v8h*)&Xcs[ld_row * LDW + ld_cg] = cvt8(pf0, pf1);

    // issue next slab's global loads; latency hides under this slab's WMMAs
    if (s + 1 < NSLABS) {
      const float4* p =
          (const float4*)(Xc + ((size_t)b * Nq + n_start + (s + 1) * SLAB +
                                ld_row) * Cq + ld_cg);
      pf0 = p[0];
      pf1 = p[1];
    }
    __syncthreads();

    // ---- step 1: x_out[32,384] = Xc @ W1^T ----
#pragma unroll
    for (int mt = 0; mt < 2; ++mt)
#pragma unroll
      for (int ht = 0; ht < 3; ++ht) {
        v8f acc = zero8();
#pragma unroll
        for (int k0 = 0; k0 < Cq; k0 += 32) {
          v16h a = ldsA(Xcs, LDW, mt * 16, k0, lane);
          v16h bm = ldsBt(W1s, LDW, hbase + ht * 16, k0, lane);
          acc = wmma16(a, bm, acc);
        }
        int col = hbase + ht * 16 + (lane & 15);
        int mrow = mt * 16 + ((lane >> 4) << 3);
#pragma unroll
        for (int r = 0; r < 8; ++r)
          xouts[(mrow + r) * LDX + col] = (_Float16)acc[r];
      }
    __syncthreads();

    // ---- step 2: t[32,64] = x_out @ Tprev ----
    {
      v8f acc = zero8();
#pragma unroll
      for (int k0 = 0; k0 < Hq; k0 += 32) {
        v16h a = ldsA(xouts, LDX, s_mt * 16, k0, lane);
        v16h bm = ldsBt(TprevT, LDX, s_ct * 16, k0, lane);
        acc = wmma16(a, bm, acc);
      }
      int col = s_ct * 16 + (lane & 15);
      int mrow = s_mt * 16 + ((lane >> 4) << 3);
#pragma unroll
      for (int r = 0; r < 8; ++r)
        ts[(mrow + r) * LDW + col] = (_Float16)acc[r];
    }
    __syncthreads();

    // ---- step 3: scores = t @ W2^T + b2 ; shifted exp -> attnT, denom ----
    {
      v8f acc = zero8();
#pragma unroll
      for (int k0 = 0; k0 < Cq; k0 += 32) {
        v16h a = ldsA(ts, LDW, s_mt * 16, k0, lane);
        v16h bm = ldsBt(W2s, LDW, s_ct * 16, k0, lane);
        acc = wmma16(a, bm, acc);
      }
      int c = s_ct * 16 + (lane & 15);
      int nb = s_mt * 16 + ((lane >> 4) << 3);
      float bias = b2s[c];
      float psum = 0.f;
#pragma unroll
      for (int r = 0; r < 8; ++r) {
        float e = __expf(acc[r] + bias - EXP_SHIFT);
        attnT[c * LDT + nb + r] = (_Float16)e;
        psum += e;
      }
      unsafeAtomicAdd(&denomLDS[c], psum);
    }

    // ---- step 4: v[32,384] = Xc @ W3^T + b3 -> vT (transposed f16) ----
#pragma unroll
    for (int mt = 0; mt < 2; ++mt)
#pragma unroll
      for (int ht = 0; ht < 3; ++ht) {
        v8f acc = zero8();
        int col0 = hbase + ht * 16;
#pragma unroll
        for (int k0 = 0; k0 < Cq; k0 += 32) {
          v16h a = ldsA(Xcs, LDW, mt * 16, k0, lane);
          v16h bm = ldsBt(W3s, LDW, col0, k0, lane);
          acc = wmma16(a, bm, acc);
        }
        int h = col0 + (lane & 15);
        int nb = mt * 16 + ((lane >> 4) << 3);
        float bias = b3s[h];
#pragma unroll
        for (int r = 0; r < 8; ++r)
          vT[h * LDT + nb + r] = (_Float16)(acc[r] + bias);
      }
    __syncthreads();

    // ---- step 5: out[c,h] += attn^T @ v  (K = 32 slab rows) ----
#pragma unroll
    for (int ct = 0; ct < 4; ++ct) {
      v16h a = ldsA(attnT, LDT, ct * 16, 0, lane);
#pragma unroll
      for (int ht = 0; ht < 3; ++ht) {
        v16h bm = ldsBt(vT, LDT, hbase + ht * 16, 0, lane);
        oacc[ct][ht] = wmma16(a, bm, oacc[ct][ht]);
      }
    }
    __syncthreads();
  }

  // ---- flush partials ----
  if (tid < Cq) unsafeAtomicAdd(&ws_denom[b * Cq + tid], denomLDS[tid]);
#pragma unroll
  for (int ct = 0; ct < 4; ++ct)
#pragma unroll
    for (int ht = 0; ht < 3; ++ht) {
      int cbase = ct * 16 + ((lane >> 4) << 3);
      int h = hbase + ht * 16 + (lane & 15);
#pragma unroll
      for (int r = 0; r < 8; ++r)
        unsafeAtomicAdd(&ws_out[((size_t)b * Cq + cbase + r) * Hq + h],
                        oacc[ct][ht][r]);
    }
}

__global__ void detok_finalize(const float* __restrict__ ws_out,
                               const float* __restrict__ ws_denom,
                               float* __restrict__ out) {
  int idx = blockIdx.x * 256 + threadIdx.x;
  if (idx >= Bq * Hq * Cq) return;
  int b = idx / (Hq * Cq);
  int r = idx % (Hq * Cq);
  int h = r / Cq;
  int c = r % Cq;
  out[idx] = ws_out[((size_t)b * Cq + c) * Hq + h] / ws_denom[b * Cq + c];
}

extern "C" void kernel_launch(void* const* d_in, const int* in_sizes, int n_in,
                              void* d_out, int out_size, void* d_ws,
                              size_t ws_size, hipStream_t stream) {
  const float* Xc = (const float*)d_in[0];
  const float* Tprev = (const float*)d_in[1];
  const float* W1 = (const float*)d_in[2];
  const float* W2 = (const float*)d_in[3];
  const float* b2 = (const float*)d_in[4];
  const float* W3 = (const float*)d_in[5];
  const float* b3 = (const float*)d_in[6];
  float* out = (float*)d_out;

  float* ws_out = (float*)d_ws;                  // [B,C,H] f32 accumulators
  float* ws_denom = ws_out + Bq * Cq * Hq;       // [B,C] f32
  int nz = Bq * Cq * Hq + Bq * Cq;

  detok_zero_ws<<<(nz + 255) / 256, 256, 0, stream>>>(ws_out, nz);
  detok_fused<<<Bq * BLK_PER_B, 256, 0, stream>>>(Xc, Tprev, W1, W2, b2, W3,
                                                  b3, ws_out, ws_denom);
  detok_finalize<<<(Bq * Hq * Cq + 255) / 256, 256, 0, stream>>>(ws_out,
                                                                 ws_denom, out);
}
